// WeightOnlyInt4Linear_42382737276954
// MI455X (gfx1250) — matrix operand order repaired and checked by reading, not compiled, not run on hardware
//
#include <hip/hip_runtime.h>
#include <hip/hip_bf16.h>
#include <math.h>

// ---------------------------------------------------------------------------
// Weight-only int4 GEMM for gfx1250 (MI455X):
//   out[M,N] = x[M,K] (bf16) @ dequant(Wq[N,K])^T,  w = (q-8)*scale + zero
// bf16 WMMA 16x16x32; double-buffered LDS; async global->LDS copy for x tile;
// weight q-values register-preloaded one stage ahead; 128x128 tile / block.
// Grid ordered so the 8 M-tile blocks sharing a weight tile are consecutive
// (weights = 180MB stream HBM once, re-hits served by the 192MB L2).
// ---------------------------------------------------------------------------

#define GROUPSIZE 128

typedef __bf16 bf16;
typedef __attribute__((ext_vector_type(16))) __bf16 bf16x16;
typedef __attribute__((ext_vector_type(8)))  float  f32x8;
typedef __attribute__((ext_vector_type(4)))  unsigned int u32x4;
typedef __attribute__((ext_vector_type(4)))  int i32x4;

constexpr int BM = 128;            // M tile per block
constexpr int BN = 128;            // N tile per block
constexpr int BK = 64;             // K slab per stage (2 WMMA k-steps)
constexpr int LDSS = BK + 8;       // padded LDS stride (bf16) -> 144B rows

__global__ __launch_bounds__(256)
void int4_wmma_gemm(const bf16* __restrict__ X,      // [M,K]
                    const int*  __restrict__ Wq,     // [N,K] values in [0,16)
                    const bf16* __restrict__ SAZ,    // [G,N,2] {scale, zero}
                    bf16* __restrict__ Out,          // [M,N]
                    int M, int N, int K)
{
    __shared__ __align__(16) bf16 As[2][BM * LDSS];  // x tiles (ping-pong)
    __shared__ __align__(16) bf16 Ws[2][BN * LDSS];  // dequant w tiles

    const int tid  = threadIdx.x;
    const int lane = tid & 31;
    const int wave = tid >> 5;          // 0..7
    const int wm   = wave & 3;          // wave's 32-row chunk of M tile
    const int wn   = wave >> 2;         // wave's 64-col chunk of N tile
    const int m0   = blockIdx.x * BM;   // x = M-tile: consecutive blocks share W
    const int n0   = blockIdx.y * BN;
    const int half = lane >> 4;
    const int r    = lane & 15;

    // per-thread staging coordinates: 4 chunks of 8 elements (coalesced b128)
    int srow[4], skc[4];
    #pragma unroll
    for (int j = 0; j < 4; ++j) {
        const int chunk = j * 256 + tid;   // 1024 chunks over the 128x64 slab
        srow[j] = chunk >> 3;              // 0..127
        skc[j]  = (chunk & 7) << 3;        // 0..56
    }

    f32x8 acc[2][4];
    #pragma unroll
    for (int i = 0; i < 2; ++i)
        #pragma unroll
        for (int j = 0; j < 4; ++j)
            acc[i][j] = (f32x8){0.f,0.f,0.f,0.f,0.f,0.f,0.f,0.f};

    i32x4 qreg[4][2];          // next-stage weight q values (reg-preloaded)
    unsigned int szreg[4];     // packed {zero.bf16, scale.bf16}

    // Issue next-stage loads: weight q -> VGPRs, x slab -> LDS via async DMA.
    auto issue_loads = [&](int k0, int buf) {
        #pragma unroll
        for (int j = 0; j < 4; ++j) {
            const int n  = n0 + srow[j];
            const int kg = k0 + skc[j];
            const int g  = kg >> 7;                       // / GROUPSIZE
            const int* qp = Wq + (size_t)n * K + kg;
            qreg[j][0] = *reinterpret_cast<const i32x4*>(qp);
            qreg[j][1] = *reinterpret_cast<const i32x4*>(qp + 4);
            szreg[j] = *reinterpret_cast<const unsigned int*>(
                           SAZ + ((size_t)g * N + n) * 2);
            // async copy: 16B of x per lane, global -> LDS (ASYNCcnt path)
            unsigned int lds_off = (unsigned int)(size_t)
                &As[buf][srow[j] * LDSS + skc[j]];
            unsigned long long ga = (unsigned long long)(const void*)
                (X + (size_t)(m0 + srow[j]) * K + kg);
            asm volatile("global_load_async_to_lds_b128 %0, %1, off"
                         :: "v"(lds_off), "v"(ga) : "memory");
        }
    };

    // Dequantize the register-held q values and store the bf16 tile to LDS.
    auto dequant_store = [&](int buf) {
        #pragma unroll
        for (int j = 0; j < 4; ++j) {
            const unsigned int u = szreg[j];
            const float scale = __builtin_bit_cast(float, u << 16);         // bf16->f32
            const float zero  = __builtin_bit_cast(float, u & 0xffff0000u); // bf16->f32
            union { bf16 h[8]; u32x4 v; } t;
            #pragma unroll
            for (int e = 0; e < 4; ++e)
                t.h[e] = (bf16)((float)(qreg[j][0][e] - 8) * scale + zero);
            #pragma unroll
            for (int e = 0; e < 4; ++e)
                t.h[4 + e] = (bf16)((float)(qreg[j][1][e] - 8) * scale + zero);
            *reinterpret_cast<u32x4*>(&Ws[buf][srow[j] * LDSS + skc[j]]) = t.v;
        }
    };

    // ---- prologue: fill buffer 0 ----
    issue_loads(0, 0);
    dequant_store(0);
    asm volatile("s_wait_asynccnt 0x0" ::: "memory");
    __syncthreads();

    const int nstages = K / BK;
    for (int s = 0; s < nstages; ++s) {
        const int buf = s & 1;
        const bool more = (s + 1 < nstages);
        if (more)
            issue_loads((s + 1) * BK, buf ^ 1);   // in flight during compute

        // ---- compute on buffer `buf`: two K=32 WMMA steps ----
        #pragma unroll
        for (int kk = 0; kk < BK; kk += 32) {
            bf16x16 afrag[2], bfrag[4];

            // A (16x32): lanes 0-15 hold K[kk..+7],[kk+16..+23];
            // lanes 16-31 hold K[kk+8..+15],[kk+24..+31]; M = lane&15.
            #pragma unroll
            for (int i = 0; i < 2; ++i) {
                const int row = wm * 32 + i * 16 + r;
                const bf16* p = &As[buf][row * LDSS + kk + half * 8];
                union { bf16x16 v; u32x4 q[2]; } u;
                u.q[0] = *reinterpret_cast<const u32x4*>(p);
                u.q[1] = *reinterpret_cast<const u32x4*>(p + 16);
                afrag[i] = u.v;
            }
            // B (32x16): column n = lane&15; lanes 0-15 K[kk..+7],[kk+16..+23];
            // lanes 16-31 K[kk+8..+15],[kk+24..+31] (same order as A).
            #pragma unroll
            for (int jn = 0; jn < 4; ++jn) {
                const int row = wn * 64 + jn * 16 + r;
                const bf16* p = &Ws[buf][row * LDSS + kk + half * 8];
                union { bf16x16 v; u32x4 q[2]; } u;
                u.q[0] = *reinterpret_cast<const u32x4*>(p);
                u.q[1] = *reinterpret_cast<const u32x4*>(p + 16);
                bfrag[jn] = u.v;
            }
            #pragma unroll
            for (int i = 0; i < 2; ++i)
                #pragma unroll
                for (int jn = 0; jn < 4; ++jn)
                    acc[i][jn] = __builtin_amdgcn_wmma_f32_16x16x32_bf16(
                        false, afrag[i], false, bfrag[jn],
                        (short)0, acc[i][jn], false, false);
        }

        if (more) {
            dequant_store(buf ^ 1);                       // VALU overlaps WMMA
            asm volatile("s_wait_asynccnt 0x0" ::: "memory");
        }
        __syncthreads();
    }

    // ---- epilogue: D (16x16 f32): VGPR e -> M = 8*half + e, N = lane&15 ----
    #pragma unroll
    for (int i = 0; i < 2; ++i) {
        #pragma unroll
        for (int jn = 0; jn < 4; ++jn) {
            const int mbase = m0 + wm * 32 + i * 16 + half * 8;
            const int n     = n0 + wn * 64 + jn * 16 + r;
            #pragma unroll
            for (int e = 0; e < 8; ++e)
                Out[(size_t)(mbase + e) * N + n] = (bf16)acc[i][jn][e];
        }
    }
}

extern "C" void kernel_launch(void* const* d_in, const int* in_sizes, int n_in,
                              void* d_out, int out_size, void* d_ws, size_t ws_size,
                              hipStream_t stream) {
    const bf16* X   = (const bf16*)d_in[0];            // [B,S,K] bf16
    const int*  Wq  = (const int*)d_in[1];             // [N,K] int32
    const bf16* SAZ = (const bf16*)d_in[2];            // [G,N,2] bf16
    bf16*       Out = (bf16*)d_out;                    // [B,S,N] bf16

    // Recover dims:  a = M*K, w = N*K, o = M*N  =>  K = sqrt(a*w/o)
    const double a = (double)in_sizes[0];
    const double w = (double)in_sizes[1];
    const double o = (double)out_size;
    const int K = (int)(sqrt(a * w / o) + 0.5);
    const int M = (int)(a / K + 0.5);
    const int N = (int)(w / K + 0.5);

    dim3 grid(M / BM, N / BN);   // M-tiles fastest: weight-tile sharers adjacent
    dim3 block(256);
    int4_wmma_gemm<<<grid, block, 0, stream>>>(X, Wq, SAZ, Out, M, N, K);
}
